// WindowedMaskedAttention_72189810311254
// MI455X (gfx1250) — compile-verified, matrix operand order checked by reading, not run on hardware
//
#include <hip/hip_runtime.h>
#include <hip/hip_bf16.h>

// ---------------- problem constants (from reference) ----------------
#define HEADS   8
#define HDIM    64
#define NTOK    256          // N
#define CDIM    512          // C
#define NBP     128          // B*P
#define ROWS    32768        // B*P*N
#define INV_TEMP 0.125f      // 64^-0.5
#define LN_EPS  1e-5f

typedef _Float16 h16;
typedef __attribute__((ext_vector_type(16))) _Float16 v16h;
typedef __attribute__((ext_vector_type(8)))  _Float16 v8h;
typedef __attribute__((ext_vector_type(8)))  float    v8f;

union FragH { v16h v; v8h h[2]; };

// Load a 16-element f16 WMMA A/B fragment for one lane.
// rowPtr points at the start of this lane's row within the 32-wide K window.
// Per ISA 7.12.2 (16-bit A 16x32): lane-group g holds K = [8g,8g+8) and [16+8g,24+8g).
__device__ inline v16h load_frag(const h16* rowPtr, int g) {
    FragH f;
    f.h[0] = *(const v8h*)(rowPtr + 8 * g);
    f.h[1] = *(const v8h*)(rowPtr + 16 + 8 * g);
    return f.v;
}

// CDNA5 async global->LDS copy (16 bytes), tracked by ASYNCcnt.
// LDS destination address = low 32 bits of the generic pointer (LDS aperture).
__device__ inline void async_copy_b128(void* ldsDst, const void* gSrc) {
    unsigned l = (unsigned)(unsigned long long)(uintptr_t)ldsDst;
    asm volatile("global_load_async_to_lds_b128 %0, %1, off"
                 :: "v"(l), "v"(gSrc) : "memory");
}
__device__ inline void wait_async0() {
    asm volatile("s_wait_asynccnt 0x0" ::: "memory");
}

// ---------------- kernel 0: f32 -> f16 weight conversion ----------------
__global__ void cvt_f32_to_h16(const float* __restrict__ src, h16* __restrict__ dst, int n) {
    int i = blockIdx.x * blockDim.x + threadIdx.x;
    if (i < n) dst[i] = (h16)src[i];
}

// ---------------- kernel 1: LayerNorm row -> f16 ----------------
__global__ __launch_bounds__(256)
void ln_to_h16(const float* __restrict__ x, const float* __restrict__ gain,
               const float* __restrict__ beta, h16* __restrict__ y) {
    __shared__ float red[256];
    const int row = blockIdx.x;
    const int t = threadIdx.x;
    const float* xr = x + (size_t)row * CDIM;
    float a0 = xr[t], a1 = xr[t + 256];

    red[t] = a0 + a1;
    __syncthreads();
    for (int off = 128; off > 0; off >>= 1) {
        if (t < off) red[t] += red[t + off];
        __syncthreads();
    }
    const float mean = red[0] * (1.0f / CDIM);
    __syncthreads();

    const float d0 = a0 - mean, d1 = a1 - mean;
    red[t] = d0 * d0 + d1 * d1;
    __syncthreads();
    for (int off = 128; off > 0; off >>= 1) {
        if (t < off) red[t] += red[t + off];
        __syncthreads();
    }
    const float rstd = rsqrtf(red[0] * (1.0f / CDIM) + LN_EPS);

    h16* yr = y + (size_t)row * CDIM;
    yr[t]       = (h16)(d0 * rstd * gain[t]       + beta[t]);
    yr[t + 256] = (h16)(d1 * rstd * gain[t + 256] + beta[t + 256]);
}

// ---------------- kernel 2: WMMA GEMM  Y[M,512] = X[M,512] @ W[512,512]^T ----------------
// Block tile 128(M) x 64(N), 8 waves (4Mx2N); each wave: 32x32 = 4 WMMA accums from
// 2 A-frags + 2 B-frags (1 fragment load per WMMA). K staged 32-wide through LDS,
// double-buffered, filled with async global->LDS copies overlapped with the WMMAs.
#define GEMM_M 128
#define GEMM_N 64
#define LDK 40   // padded K-stride (f16 elems), keeps 16B alignment
__global__ __launch_bounds__(256)
void gemm_h16(const h16* __restrict__ X, const h16* __restrict__ W,
              const float* __restrict__ bias, float* __restrict__ Yf,
              h16* __restrict__ Yh) {
    __shared__ h16 Xs[2][GEMM_M * LDK];
    __shared__ h16 Ws[2][GEMM_N * LDK];

    const int t    = threadIdx.x;
    const int wv   = t >> 5;          // wave id (wave32)
    const int lane = t & 31;
    const int g    = lane >> 4;       // lane group
    const int ml   = lane & 15;
    const int mw   = wv >> 1;         // 4 waves along M
    const int nw   = wv & 1;          // 2 waves along N
    const int bm   = blockIdx.x * GEMM_M;
    const int bn   = blockIdx.y * GEMM_N;
    // coop-load mapping: X tile 128 rows x 32 cols (2x16B per thread),
    //                    W tile 64 rows x 32 cols (1x16B per thread)
    const int xrow = t >> 1, xcol = (t & 1) * 16;
    const int wrow = t >> 2, wcol = (t & 3) * 8;

    v8f acc00 = {}, acc01 = {}, acc10 = {}, acc11 = {};

    // prologue: stage K-slice 0 into buffer 0
    async_copy_b128(&Xs[0][xrow * LDK + xcol],     X + (size_t)(bm + xrow) * CDIM + xcol);
    async_copy_b128(&Xs[0][xrow * LDK + xcol + 8], X + (size_t)(bm + xrow) * CDIM + xcol + 8);
    async_copy_b128(&Ws[0][wrow * LDK + wcol],     W + (size_t)(bn + wrow) * CDIM + wcol);
    wait_async0();
    __syncthreads();

    for (int k0 = 0; k0 < CDIM; k0 += 32) {
        const int cur = (k0 >> 5) & 1;
        if (k0 + 32 < CDIM) {   // stage next K-slice into the other buffer (async)
            const int nk = k0 + 32;
            async_copy_b128(&Xs[cur ^ 1][xrow * LDK + xcol],     X + (size_t)(bm + xrow) * CDIM + nk + xcol);
            async_copy_b128(&Xs[cur ^ 1][xrow * LDK + xcol + 8], X + (size_t)(bm + xrow) * CDIM + nk + xcol + 8);
            async_copy_b128(&Ws[cur ^ 1][wrow * LDK + wcol],     W + (size_t)(bn + wrow) * CDIM + nk + wcol);
        }
        v16h a0 = load_frag(&Xs[cur][(mw * 32 +      ml) * LDK], g);
        v16h a1 = load_frag(&Xs[cur][(mw * 32 + 16 + ml) * LDK], g);
        v16h b0 = load_frag(&Ws[cur][(nw * 32 +      ml) * LDK], g);
        v16h b1 = load_frag(&Ws[cur][(nw * 32 + 16 + ml) * LDK], g);
        acc00 = __builtin_amdgcn_wmma_f32_16x16x32_f16(false, a0, false, b0, (short)0, acc00, false, false);
        acc01 = __builtin_amdgcn_wmma_f32_16x16x32_f16(false, a0, false, b1, (short)0, acc01, false, false);
        acc10 = __builtin_amdgcn_wmma_f32_16x16x32_f16(false, a1, false, b0, (short)0, acc10, false, false);
        acc11 = __builtin_amdgcn_wmma_f32_16x16x32_f16(false, a1, false, b1, (short)0, acc11, false, false);
        wait_async0();
        __syncthreads();
    }

    const int nb = bn + nw * 32 + ml;
#pragma unroll
    for (int r = 0; r < 8; ++r) {
        const int m0 = bm + mw * 32 + r + 8 * g;   // C/D layout: M = r + 8*(lane/16)
        const int m1 = m0 + 16;
        if (Yf) {
            const float bb0 = bias[nb], bb1 = bias[nb + 16];
            Yf[(size_t)m0 * CDIM + nb]      = acc00[r] + bb0;
            Yf[(size_t)m0 * CDIM + nb + 16] = acc01[r] + bb1;
            Yf[(size_t)m1 * CDIM + nb]      = acc10[r] + bb0;
            Yf[(size_t)m1 * CDIM + nb + 16] = acc11[r] + bb1;
        } else {
            Yh[(size_t)m0 * CDIM + nb]      = (h16)acc00[r];
            Yh[(size_t)m0 * CDIM + nb + 16] = (h16)acc01[r];
            Yh[(size_t)m1 * CDIM + nb]      = (h16)acc10[r];
            Yh[(size_t)m1 * CDIM + nb + 16] = (h16)acc11[r];
        }
    }
}

// ---------------- kernel 3: windowed masked attention per (b*p, h) ----------------
// LDS: Sh 256x256 f16 (128KB) | Qs 256x64 f16 (32KB) | Ks 256x64 f16 -> reused as Vt 64x256 (32KB)
#define SMEM_ATTN (131072 + 32768 + 32768)
__global__ __launch_bounds__(256)
void attn_h16(const h16* __restrict__ qh, const h16* __restrict__ kh,
              const h16* __restrict__ vh, const float* __restrict__ mask,
              const float* __restrict__ pos, h16* __restrict__ oh) {
    extern __shared__ char smem[];
    h16* Sh = (h16*)smem;                      // scores then probabilities, stride 256
    h16* Qs = (h16*)(smem + 131072);           // 256 x 64
    h16* Ks = (h16*)(smem + 131072 + 32768);   // 256 x 64 (later Vt: 64 x 256)

    const int bp   = blockIdx.x;
    const int h    = blockIdx.y;
    const int t    = threadIdx.x;
    const int wv   = t >> 5;
    const int lane = t & 31;
    const int g    = lane >> 4;
    const int nl   = lane & 15;
    const size_t rowbase = (size_t)bp * NTOK;

    const float* maskb = mask + (size_t)bp * NTOK * NTOK;
    const float* posb  = pos  + (size_t)h  * NTOK * NTOK;
    __builtin_prefetch(maskb + (size_t)t * NTOK, 0, 1);   // global_prefetch_b8

    // stage Q and K tiles via async global->LDS (each thread: one 64-elem row each)
    {
        const h16* qg = qh + (rowbase + t) * CDIM + h * HDIM;
        const h16* kg = kh + (rowbase + t) * CDIM + h * HDIM;
#pragma unroll
        for (int c = 0; c < 8; ++c) {
            async_copy_b128(Qs + t * HDIM + c * 8, qg + c * 8);
            async_copy_b128(Ks + t * HDIM + c * 8, kg + c * 8);
        }
        wait_async0();
    }
    __syncthreads();

    // Phase 1: S = mask * (Q K^T * inv_temp + pos)   (256 WMMA tiles / 8 waves)
    for (int idx = wv; idx < 256; idx += 8) {
        const int ti = idx >> 4, tj = idx & 15;
        v8f acc = {};
#pragma unroll
        for (int k0 = 0; k0 < HDIM; k0 += 32) {
            v16h a = load_frag(Qs + (ti * 16 + nl) * HDIM + k0, g);
            v16h b = load_frag(Ks + (tj * 16 + nl) * HDIM + k0, g);
            acc = __builtin_amdgcn_wmma_f32_16x16x32_f16(false, a, false, b, (short)0, acc, false, false);
        }
        const int j = tj * 16 + nl;
#pragma unroll
        for (int r = 0; r < 8; ++r) {
            const int i = ti * 16 + r + 8 * g;
            float s = acc[r] * INV_TEMP + posb[i * NTOK + j];
            s *= maskb[i * NTOK + j];
            Sh[i * NTOK + j] = (h16)s;
        }
    }
    __syncthreads();

    // stage V transposed into Ks region: Vt[d][j], stride NTOK
    {
        const h16* vg = vh + (rowbase + t) * CDIM + h * HDIM;
        h16 vrow[HDIM];
#pragma unroll
        for (int c = 0; c < 8; ++c) *(v8h*)(vrow + c * 8) = *(const v8h*)(vg + c * 8);
#pragma unroll
        for (int d = 0; d < HDIM; ++d) Ks[d * NTOK + t] = vrow[d];
    }

    // Phase 2: softmax, one row per thread, f32 math, in-place f16 rewrite
    {
        h16* sr = Sh + t * NTOK;
        float mx = -3.0e38f;
        for (int j = 0; j < NTOK; ++j) mx = fmaxf(mx, (float)sr[j]);
        float sum = 0.0f;
        for (int j = 0; j < NTOK; ++j) sum += __expf((float)sr[j] - mx);
        const float inv = 1.0f / sum;
        for (int j = 0; j < NTOK; ++j) sr[j] = (h16)(__expf((float)sr[j] - mx) * inv);
    }
    __syncthreads();

    // Phase 3: O = P(256x256) @ V(256x64)   (64 WMMA tiles / 8 waves)
    for (int idx = wv; idx < 64; idx += 8) {
        const int ti = idx >> 2, tn = idx & 3;
        v8f acc = {};
#pragma unroll
        for (int k0 = 0; k0 < NTOK; k0 += 32) {
            v16h a = load_frag(Sh + (ti * 16 + nl) * NTOK + k0, g);
            v16h b = load_frag(Ks + (tn * 16 + nl) * NTOK + k0, g);  // Vt, K-contiguous
            acc = __builtin_amdgcn_wmma_f32_16x16x32_f16(false, a, false, b, (short)0, acc, false, false);
        }
        const int d = tn * 16 + nl;
#pragma unroll
        for (int r = 0; r < 8; ++r) {
            const int i = ti * 16 + r + 8 * g;
            oh[(rowbase + i) * CDIM + h * HDIM + d] = (h16)acc[r];
        }
    }
}

// ---------------- host launch ----------------
extern "C" void kernel_launch(void* const* d_in, const int* in_sizes, int n_in,
                              void* d_out, int out_size, void* d_ws, size_t ws_size,
                              hipStream_t stream) {
    const float* q      = (const float*)d_in[0];
    const float* k      = (const float*)d_in[1];
    const float* v      = (const float*)d_in[2];
    const float* mask   = (const float*)d_in[3];
    const float* pos    = (const float*)d_in[4];
    const float* ln_q_g = (const float*)d_in[5];
    const float* ln_q_b = (const float*)d_in[6];
    const float* ln_k_g = (const float*)d_in[7];
    const float* ln_k_b = (const float*)d_in[8];
    const float* ln_v_g = (const float*)d_in[9];
    const float* ln_v_b = (const float*)d_in[10];
    const float* Wq     = (const float*)d_in[11];
    const float* Wk     = (const float*)d_in[12];
    const float* Wv     = (const float*)d_in[13];
    const float* Wo     = (const float*)d_in[14];
    const float* bo     = (const float*)d_in[15];
    float* out = (float*)d_out;

    // workspace carve-up (f16 buffers)
    char* wsb = (char*)d_ws;
    const size_t WSZ = (size_t)CDIM * CDIM * sizeof(h16);   // 512 KB per weight
    h16* wqb = (h16*)(wsb + 0 * WSZ);
    h16* wkb = (h16*)(wsb + 1 * WSZ);
    h16* wvb = (h16*)(wsb + 2 * WSZ);
    h16* wob = (h16*)(wsb + 3 * WSZ);
    const size_t RSZ = (size_t)ROWS * CDIM * sizeof(h16);   // 32 MB per activation
    char* act = wsb + 4 * WSZ;
    h16* xq = (h16*)(act + 0 * RSZ);
    h16* xk = (h16*)(act + 1 * RSZ);
    h16* xv = (h16*)(act + 2 * RSZ);
    h16* qh = (h16*)(act + 3 * RSZ);
    h16* kh = (h16*)(act + 4 * RSZ);
    h16* vh = (h16*)(act + 5 * RSZ);
    h16* oh = xq;   // xq dead after its projection; attention runs later

    // weights -> f16
    const int wn = CDIM * CDIM;
    cvt_f32_to_h16<<<(wn + 255) / 256, 256, 0, stream>>>(Wq, wqb, wn);
    cvt_f32_to_h16<<<(wn + 255) / 256, 256, 0, stream>>>(Wk, wkb, wn);
    cvt_f32_to_h16<<<(wn + 255) / 256, 256, 0, stream>>>(Wv, wvb, wn);
    cvt_f32_to_h16<<<(wn + 255) / 256, 256, 0, stream>>>(Wo, wob, wn);

    // LayerNorm
    ln_to_h16<<<ROWS, 256, 0, stream>>>(q, ln_q_g, ln_q_b, xq);
    ln_to_h16<<<ROWS, 256, 0, stream>>>(k, ln_k_g, ln_k_b, xk);
    ln_to_h16<<<ROWS, 256, 0, stream>>>(v, ln_v_g, ln_v_b, xv);

    // projections (WMMA, async-staged double-buffered LDS)
    dim3 gg(ROWS / GEMM_M, CDIM / GEMM_N);
    gemm_h16<<<gg, 256, 0, stream>>>(xq, wqb, nullptr, nullptr, qh);
    gemm_h16<<<gg, 256, 0, stream>>>(xk, wkb, nullptr, nullptr, kh);
    gemm_h16<<<gg, 256, 0, stream>>>(xv, wvb, nullptr, nullptr, vh);

    // attention (WMMA + LDS softmax)
    attn_h16<<<dim3(NBP, HEADS), 256, SMEM_ATTN, stream>>>(qh, kh, vh, mask, pos, oh);

    // output projection (WMMA, f32 out + bias)
    gemm_h16<<<gg, 256, 0, stream>>>(oh, wob, bo, out, nullptr);
}
